// _KnowledgeConsistentNonLocalBlockND_30923764531427
// MI455X (gfx1250) — matrix operand order repaired and checked by reading, not compiled, not run on hardware
//
#include <hip/hip_runtime.h>
#include <hip/hip_bf16.h>

// ---------------------------------------------------------------------------
// Non-local attention block for MI455X (gfx1250, wave32, WMMA f16 16x16x32).
// All matmuls run through v_wmma_f32_16x16x32_f16 with f32 accumulation.
// LDS tile layouts are chosen so every WMMA fragment is two ds_load_b128.
// ---------------------------------------------------------------------------

#define BB   8
#define CC   256
#define CI   128
#define NN   4096   // H*W
#define MM   1024   // (H/2)*(W/2)
#define BN_EPS 1e-5f

typedef __attribute__((ext_vector_type(16))) _Float16 v16h;
typedef __attribute__((ext_vector_type(8)))  _Float16 v8h;
typedef __attribute__((ext_vector_type(4)))  _Float16 v4h;
typedef __attribute__((ext_vector_type(8)))  float    v8f;

__device__ __forceinline__ float sigmoidf_(float v) {
    return 1.0f / (1.0f + __expf(-v));
}

__device__ __forceinline__ v8f wmma_f16(v16h a, v16h b, v8f c) {
    return __builtin_amdgcn_wmma_f32_16x16x32_f16(
        false, a, false, b, (short)0, c, false, false);
}

// A fragment from LDS tile stored [m][k] (rows x 32 k, lda multiple of 8).
// Lane L: m = row_base + (L&15); lanes 0-15 take K {0..7,16..23},
// lanes 16-31 take K {8..15,24..31}. Two 16B contiguous runs -> 2x b128.
__device__ __forceinline__ v16h load_a(const _Float16* As, int lda,
                                       int row_base, int lane) {
    const int m = row_base + (lane & 15);
    const int h = (lane >> 4) << 3;  // 0 or 8
    const _Float16* p = As + m * lda;
    const v8h lo = *(const v8h*)(p + h);
    const v8h hi = *(const v8h*)(p + 16 + h);
    return __builtin_shufflevector(lo, hi, 0, 1, 2, 3, 4, 5, 6, 7,
                                           8, 9, 10, 11, 12, 13, 14, 15);
}

// B fragment from LDS tile stored [n][k] (cols x 32 k, ldb multiple of 8).
// Lane L: n = col_base + (L&15); lanes 0-15 take K {0..7,16..23},
// lanes 16-31 take K {8..15,24..31}. Two 16B contiguous runs -> 2x b128.
__device__ __forceinline__ v16h load_b(const _Float16* Bs, int ldb,
                                       int col_base, int lane) {
    const int n = col_base + (lane & 15);
    const int kh = (lane >> 4) << 3;  // 0 or 8
    const _Float16* p = Bs + n * ldb + kh;
    const v8h lo = *(const v8h*)(p);
    const v8h hi = *(const v8h*)(p + 16);
    return __builtin_shufflevector(lo, hi, 0, 1, 2, 3, 4, 5, 6, 7,
                                           8, 9, 10, 11, 12, 13, 14, 15);
}

// 16x16 f32 C/D tile scatter: lane L: col = L&15, rows v + 8*(L>>4).
__device__ __forceinline__ void store_tile_f32(float* dst, size_t row_stride,
                                               v8f c, int lane) {
    const int col = lane & 15;
    const int h = (lane >> 4) << 3;
#pragma unroll
    for (int v = 0; v < 8; ++v)
        dst[(size_t)(h + v) * row_stride + col] = c[v];
}

__device__ __forceinline__ void store_tile_f16(_Float16* dst, size_t row_stride,
                                               v8f c, int lane) {
    const int col = lane & 15;
    const int h = (lane >> 4) << 3;
#pragma unroll
    for (int v = 0; v < 8; ++v)
        dst[(size_t)(h + v) * row_stride + col] = (_Float16)c[v];
}

// ---------------------------------------------------------------------------
// Kernel 1: 1x1-conv projections (theta / g / phi) with fused 2x2 maxpool
// for g and phi. Workgroup tile 128(n) x 64(ci); 8 waves = 4x2 of 32x32.
// Outputs: theta (B,N,Ci); g (B,Ci,M); phi (B,M,Ci)  -- all f16.
// ---------------------------------------------------------------------------
__global__ __launch_bounds__(256) void proj_kernel(
    const float* __restrict__ x,
    const float* __restrict__ gw, const float* __restrict__ gb,
    const float* __restrict__ tw, const float* __restrict__ tb,
    const float* __restrict__ pw, const float* __restrict__ pb,
    _Float16* __restrict__ theta, _Float16* __restrict__ gbuf,
    _Float16* __restrict__ phibuf)
{
    __shared__ __align__(16) _Float16 As[128][40];  // [n][k]
    __shared__ __align__(16) _Float16 Bs[64][40];   // [ci][k]
    __shared__ __align__(16) float Outs[128][68];   // conv result staging

    const int t = threadIdx.x;
    const int lane = t & 31, wave = t >> 5;
    const int n0 = blockIdx.x * 128;
    const int ci0 = blockIdx.y * 64;
    const int b = blockIdx.z / 3, proj = blockIdx.z % 3;

    const float* w; const float* bias;
    if (proj == 0)      { w = tw; bias = tb; }
    else if (proj == 1) { w = gw; bias = gb; }
    else                { w = pw; bias = pb; }

    const v8f zero = {0.f,0.f,0.f,0.f,0.f,0.f,0.f,0.f};
    v8f acc[2][2];
#pragma unroll
    for (int i = 0; i < 2; ++i)
#pragma unroll
        for (int j = 0; j < 2; ++j) acc[i][j] = zero;

    const int wr = (wave >> 1) * 32;
    const int wc = (wave & 1) * 32;

    for (int kc = 0; kc < CC; kc += 32) {
        {   // As[n][c] transpose fill: 4c x 4n micro-block per thread.
            // 4 contiguous float4 reads (along n), 4 packed ds_store_b64.
            const int c4 = (t & 7) * 4;          // 0..28
            const int n4 = (t >> 3) * 4;         // 0..124
            float4 vx[4];
#pragma unroll
            for (int cc = 0; cc < 4; ++cc)
                vx[cc] = *(const float4*)(x +
                    ((size_t)(b * CC + kc + c4 + cc) * NN) + n0 + n4);
#pragma unroll
            for (int nn = 0; nn < 4; ++nn) {
                v4h pk;
                pk[0] = (_Float16)((&vx[0].x)[nn]);
                pk[1] = (_Float16)((&vx[1].x)[nn]);
                pk[2] = (_Float16)((&vx[2].x)[nn]);
                pk[3] = (_Float16)((&vx[3].x)[nn]);
                *(v4h*)&As[n4 + nn][c4] = pk;
            }
        }
        {   // Bs[o][c] = w[ci0+o][kc+c]: contiguous run -> vector store.
            const int o = t >> 2, q = t & 3;
            const float* src = w + (size_t)(ci0 + o) * CC + kc + q * 8;
            v8h vv;
#pragma unroll
            for (int i = 0; i < 8; ++i) vv[i] = (_Float16)src[i];
            *(v8h*)&Bs[o][q * 8] = vv;
        }
        __syncthreads();
        v16h af[2], bf[2];
#pragma unroll
        for (int i = 0; i < 2; ++i) af[i] = load_a(&As[0][0], 40, wr + i * 16, lane);
#pragma unroll
        for (int j = 0; j < 2; ++j) bf[j] = load_b(&Bs[0][0], 40, wc + j * 16, lane);
#pragma unroll
        for (int i = 0; i < 2; ++i)
#pragma unroll
            for (int j = 0; j < 2; ++j)
                acc[i][j] = wmma_f16(af[i], bf[j], acc[i][j]);
        __syncthreads();
    }

    // Bias + stage in LDS (f32).
    {
        const int col = lane & 15;
        const int h = (lane >> 4) << 3;
#pragma unroll
        for (int i = 0; i < 2; ++i)
#pragma unroll
            for (int j = 0; j < 2; ++j) {
                const float bv = bias[ci0 + wc + j * 16 + col];
#pragma unroll
                for (int v = 0; v < 8; ++v)
                    Outs[wr + i * 16 + h + v][wc + j * 16 + col] = acc[i][j][v] + bv;
            }
    }
    __syncthreads();

    if (proj == 0) {
        // theta: (B, N, Ci) f16 row-major; packed v8h stores.
        const int r = t >> 1, half = t & 1;
        _Float16* dst = theta + ((size_t)(b * NN + n0 + r)) * CI + ci0 + half * 32;
#pragma unroll
        for (int k = 0; k < 4; ++k) {
            const float4 lo = *(const float4*)&Outs[r][half * 32 + k * 8];
            const float4 hi = *(const float4*)&Outs[r][half * 32 + k * 8 + 4];
            v8h pk;
            pk[0] = (_Float16)lo.x; pk[1] = (_Float16)lo.y;
            pk[2] = (_Float16)lo.z; pk[3] = (_Float16)lo.w;
            pk[4] = (_Float16)hi.x; pk[5] = (_Float16)hi.y;
            pk[6] = (_Float16)hi.z; pk[7] = (_Float16)hi.w;
            *(v8h*)(dst + k * 8) = pk;
        }
    } else {
        // 128 consecutive n == 2 full image rows -> pools to 32 m values.
        const int mbase = (n0 >> 7) * 32;
        if (proj == 1) {
            // g: (B, Ci, M) f16; 8 pooled values contiguous in m -> one v8h.
            const int col = t >> 2, q = t & 3;
            v8h pk;
#pragma unroll
            for (int i = 0; i < 8; ++i) {
                const int p = q * 8 + i;
                float v0 = fmaxf(Outs[2 * p][col],      Outs[2 * p + 1][col]);
                float v1 = fmaxf(Outs[64 + 2 * p][col], Outs[64 + 2 * p + 1][col]);
                pk[i] = (_Float16)fmaxf(v0, v1);
            }
            *(v8h*)(gbuf + ((size_t)(b * CI + ci0 + col)) * MM + mbase + q * 8) = pk;
        } else {
            // phi: (B, M, Ci) f16; 8 pooled values contiguous in ci -> one v8h.
            const int p = t >> 3, j = t & 7;
            v8h pk;
#pragma unroll
            for (int i = 0; i < 8; ++i) {
                const int col = j * 8 + i;
                float v0 = fmaxf(Outs[2 * p][col],      Outs[2 * p + 1][col]);
                float v1 = fmaxf(Outs[64 + 2 * p][col], Outs[64 + 2 * p + 1][col]);
                pk[i] = (_Float16)fmaxf(v0, v1);
            }
            *(v8h*)(phibuf + ((size_t)(b * MM + mbase + p)) * CI + ci0 + j * 8) = pk;
        }
    }
}

// ---------------------------------------------------------------------------
// Kernel 2: f[n][m] = sum_ci theta[n][ci] * phi[m][ci] -> (B, N, M) f32.
// Workgroup tile 128x128, waves 4x2 of 32x64, K=128 in chunks of 32.
// ---------------------------------------------------------------------------
__global__ __launch_bounds__(256) void f_gemm_kernel(
    const _Float16* __restrict__ theta, const _Float16* __restrict__ phi,
    float* __restrict__ f)
{
    __shared__ __align__(16) _Float16 As[128][40];  // [n][ci]
    __shared__ __align__(16) _Float16 Bs[128][40];  // [m][ci]

    const int t = threadIdx.x;
    const int lane = t & 31, wave = t >> 5;
    const int m0 = blockIdx.x * 128;
    const int n0 = blockIdx.y * 128;
    const int b = blockIdx.z;

    const v8f zero = {0.f,0.f,0.f,0.f,0.f,0.f,0.f,0.f};
    v8f acc[2][4];
#pragma unroll
    for (int i = 0; i < 2; ++i)
#pragma unroll
        for (int j = 0; j < 4; ++j) acc[i][j] = zero;

    const int wr = (wave >> 1) * 32;
    const int wc = (wave & 1) * 64;

    for (int kc = 0; kc < CI; kc += 32) {
        {
            const int n = t >> 1, half = t & 1;
            const uint4* src = (const uint4*)(theta +
                ((size_t)(b * NN + n0 + n)) * CI + kc + half * 16);
            *(uint4*)&As[n][half * 16]     = src[0];
            *(uint4*)&As[n][half * 16 + 8] = src[1];
        }
        {
            const int m = t >> 1, half = t & 1;
            const uint4* src = (const uint4*)(phi +
                ((size_t)(b * MM + m0 + m)) * CI + kc + half * 16);
            *(uint4*)&Bs[m][half * 16]     = src[0];
            *(uint4*)&Bs[m][half * 16 + 8] = src[1];
        }
        __syncthreads();
        v16h af[2], bf[4];
#pragma unroll
        for (int i = 0; i < 2; ++i) af[i] = load_a(&As[0][0], 40, wr + i * 16, lane);
#pragma unroll
        for (int j = 0; j < 4; ++j) bf[j] = load_b(&Bs[0][0], 40, wc + j * 16, lane);
#pragma unroll
        for (int i = 0; i < 2; ++i)
#pragma unroll
            for (int j = 0; j < 4; ++j)
                acc[i][j] = wmma_f16(af[i], bf[j], acc[i][j]);
        __syncthreads();
    }

#pragma unroll
    for (int i = 0; i < 2; ++i)
#pragma unroll
        for (int j = 0; j < 4; ++j) {
            float* dst = f + ((size_t)(b * NN + n0 + wr + i * 16)) * MM
                           + m0 + wc + j * 16;
            store_tile_f32(dst, MM, acc[i][j], lane);
        }
}

// ---------------------------------------------------------------------------
// Kernel 3: 3x3 avg-pool (count_include_pad=False) over the (N, M) plane,
// blended with sigmoid(lamb_smooth).  f stays L2-resident (134MB < 192MB).
// ---------------------------------------------------------------------------
__global__ __launch_bounds__(256) void smooth_kernel(
    const float* __restrict__ f, const float* __restrict__ lamb_smooth,
    float* __restrict__ fb)
{
    const size_t i = (size_t)blockIdx.x * 256 + threadIdx.x;
    const int m = (int)(i & (MM - 1));
    const size_t r = i >> 10;            // b*N + n
    const int n = (int)(r & (NN - 1));
    const int b = (int)(r >> 12);

    const float ls = sigmoidf_(lamb_smooth[0]);
    float s = 0.0f;
    int cnt = 0;
#pragma unroll
    for (int dn = -1; dn <= 1; ++dn) {
        const int nn = n + dn;
        if (nn < 0 || nn >= NN) continue;
        const float* rp = f + ((size_t)b * NN + nn) * MM;
#pragma unroll
        for (int dm = -1; dm <= 1; ++dm) {
            const int mm2 = m + dm;
            if (mm2 < 0 || mm2 >= MM) continue;
            s += rp[mm2];
            ++cnt;
        }
    }
    fb[i] = ls * f[i] + (1.0f - ls) * (s / (float)cnt);
}

// ---------------------------------------------------------------------------
// Kernel 4: per-row softmax over M + recurrent temporal blend, emit f16.
// One 256-thread block per (b, n) row (1024 elems, 4 per thread).
// ---------------------------------------------------------------------------
__global__ __launch_bounds__(256) void softmax_kernel(
    const float* __restrict__ fb, const float* __restrict__ nl_prev,
    const float* __restrict__ mask_prev, const float* __restrict__ lamb,
    _Float16* __restrict__ fh)
{
    __shared__ float red[256];
    const int r = blockIdx.x;            // b*N + n
    const int t = threadIdx.x;
    const float* row = fb + (size_t)r * MM;

    float v[4];
#pragma unroll
    for (int i = 0; i < 4; ++i) v[i] = row[t + i * 256];

    float mx = fmaxf(fmaxf(v[0], v[1]), fmaxf(v[2], v[3]));
    red[t] = mx;
    __syncthreads();
    for (int s = 128; s > 0; s >>= 1) {
        if (t < s) red[t] = fmaxf(red[t], red[t + s]);
        __syncthreads();
    }
    mx = red[0];
    __syncthreads();

    float e[4], sum = 0.0f;
#pragma unroll
    for (int i = 0; i < 4; ++i) { e[i] = __expf(v[i] - mx); sum += e[i]; }
    red[t] = sum;
    __syncthreads();
    for (int s = 128; s > 0; s >>= 1) {
        if (t < s) red[t] += red[t + s];
        __syncthreads();
    }
    const float inv = 1.0f / red[0];

    const float l = sigmoidf_(lamb[0]);
    const float mp = mask_prev[r];       // (B,1,H,W) flat == b*N+n
    const float denom = 1.0f / (l + (1.0f - l) * mp);
    const float* prow = nl_prev + (size_t)r * MM;
    _Float16* out = fh + (size_t)r * MM;
#pragma unroll
    for (int i = 0; i < 4; ++i) {
        const float fdc = e[i] * inv;
        const float o = (l * fdc + (1.0f - l) * prow[t + i * 256] * mp) * denom;
        out[t + i * 256] = (_Float16)o;
    }
}

// ---------------------------------------------------------------------------
// Kernel 5: y = f_div_C (N x M) * g (M x Ci) -> (B, N, Ci) f16.
// g is stored (B, Ci, M) so B-tiles [ci][m] fill with vector copies.
// Workgroup tile 128x64, waves 4x2 of 32x32, K=1024 in chunks of 32.
// ---------------------------------------------------------------------------
__global__ __launch_bounds__(256) void y_gemm_kernel(
    const _Float16* __restrict__ fh, const _Float16* __restrict__ gbuf,
    _Float16* __restrict__ y)
{
    __shared__ __align__(16) _Float16 As[128][40];  // [n][m-chunk]
    __shared__ __align__(16) _Float16 Bs[64][40];   // [ci][m-chunk]

    const int t = threadIdx.x;
    const int lane = t & 31, wave = t >> 5;
    const int ci0 = blockIdx.x * 64;
    const int n0 = blockIdx.y * 128;
    const int b = blockIdx.z;

    const v8f zero = {0.f,0.f,0.f,0.f,0.f,0.f,0.f,0.f};
    v8f acc[2][2];
#pragma unroll
    for (int i = 0; i < 2; ++i)
#pragma unroll
        for (int j = 0; j < 2; ++j) acc[i][j] = zero;

    const int wr = (wave >> 1) * 32;
    const int wc = (wave & 1) * 32;

    for (int kc = 0; kc < MM; kc += 32) {
        {
            const int n = t >> 1, half = t & 1;
            const uint4* src = (const uint4*)(fh +
                ((size_t)(b * NN + n0 + n)) * MM + kc + half * 16);
            *(uint4*)&As[n][half * 16]     = src[0];
            *(uint4*)&As[n][half * 16 + 8] = src[1];
        }
        {
            const int ci = t >> 2, q = t & 3;
            const uint4* src = (const uint4*)(gbuf +
                ((size_t)(b * CI + ci0 + ci)) * MM + kc + q * 8);
            *(uint4*)&Bs[ci][q * 8] = src[0];
        }
        __syncthreads();
        v16h af[2], bf[2];
#pragma unroll
        for (int i = 0; i < 2; ++i) af[i] = load_a(&As[0][0], 40, wr + i * 16, lane);
#pragma unroll
        for (int j = 0; j < 2; ++j) bf[j] = load_b(&Bs[0][0], 40, wc + j * 16, lane);
#pragma unroll
        for (int i = 0; i < 2; ++i)
#pragma unroll
            for (int j = 0; j < 2; ++j)
                acc[i][j] = wmma_f16(af[i], bf[j], acc[i][j]);
        __syncthreads();
    }

#pragma unroll
    for (int i = 0; i < 2; ++i)
#pragma unroll
        for (int j = 0; j < 2; ++j) {
            _Float16* dst = y + ((size_t)(b * NN + n0 + wr + i * 16)) * CI
                              + ci0 + wc + j * 16;
            store_tile_f16(dst, CI, acc[i][j], lane);
        }
}

// ---------------------------------------------------------------------------
// Kernel 6: W_y = y (N x Ci) * W_w^T (Ci x C) + W_b -> (B, C, N) f32,
// plus per-channel sum / sumsq partial reductions for BatchNorm.
// ---------------------------------------------------------------------------
__global__ __launch_bounds__(256) void wy_gemm_kernel(
    const _Float16* __restrict__ y, const float* __restrict__ Ww,
    const float* __restrict__ Wb, float* __restrict__ Wy,
    float* __restrict__ stats)
{
    __shared__ __align__(16) _Float16 As[128][40];  // [n][ci-chunk]
    __shared__ __align__(16) _Float16 Bs[64][40];   // [c][ci-chunk]
    __shared__ __align__(16) float Outs[128][68];
    __shared__ float ps[4][68];
    __shared__ float ps2[4][68];

    const int t = threadIdx.x;
    const int lane = t & 31, wave = t >> 5;
    const int c0 = blockIdx.x * 64;
    const int n0 = blockIdx.y * 128;
    const int b = blockIdx.z;

    const v8f zero = {0.f,0.f,0.f,0.f,0.f,0.f,0.f,0.f};
    v8f acc[2][2];
#pragma unroll
    for (int i = 0; i < 2; ++i)
#pragma unroll
        for (int j = 0; j < 2; ++j) acc[i][j] = zero;

    const int wr = (wave >> 1) * 32;
    const int wc = (wave & 1) * 32;

    for (int kc = 0; kc < CI; kc += 32) {
        {
            const int n = t >> 1, half = t & 1;
            const uint4* src = (const uint4*)(y +
                ((size_t)(b * NN + n0 + n)) * CI + kc + half * 16);
            *(uint4*)&As[n][half * 16]     = src[0];
            *(uint4*)&As[n][half * 16 + 8] = src[1];
        }
        {   // Bs[c][ci] = Ww[c0+c][kc+ci]  (contiguous, f32->f16, vector store)
            const int c = t >> 2, q = t & 3;
            const float* src = Ww + (size_t)(c0 + c) * CI + kc + q * 8;
            v8h vv;
#pragma unroll
            for (int i = 0; i < 8; ++i) vv[i] = (_Float16)src[i];
            *(v8h*)&Bs[c][q * 8] = vv;
        }
        __syncthreads();
        v16h af[2], bf[2];
#pragma unroll
        for (int i = 0; i < 2; ++i) af[i] = load_a(&As[0][0], 40, wr + i * 16, lane);
#pragma unroll
        for (int j = 0; j < 2; ++j) bf[j] = load_b(&Bs[0][0], 40, wc + j * 16, lane);
#pragma unroll
        for (int i = 0; i < 2; ++i)
#pragma unroll
            for (int j = 0; j < 2; ++j)
                acc[i][j] = wmma_f16(af[i], bf[j], acc[i][j]);
        __syncthreads();
    }

    {   // bias + stage
        const int col = lane & 15;
        const int h = (lane >> 4) << 3;
#pragma unroll
        for (int i = 0; i < 2; ++i)
#pragma unroll
            for (int j = 0; j < 2; ++j) {
                const float bv = Wb[c0 + wc + j * 16 + col];
#pragma unroll
                for (int v = 0; v < 8; ++v)
                    Outs[wr + i * 16 + h + v][wc + j * 16 + col] = acc[i][j][v] + bv;
            }
    }
    __syncthreads();

    // Coalesced (B, C, N) store + per-channel partial stats.
    {
        const int col = t >> 2, q = t & 3;
        float* dst = Wy + ((size_t)(b * CC + c0 + col)) * NN + n0 + q * 32;
        float s = 0.0f, s2 = 0.0f;
#pragma unroll
        for (int rr = 0; rr < 32; ++rr) {
            const float v = Outs[q * 32 + rr][col];
            dst[rr] = v;
            s += v;
            s2 += v * v;
        }
        ps[q][col] = s;
        ps2[q][col] = s2;
    }
    __syncthreads();
    {
        const int col = t >> 2, q = t & 3;
        if (q == 0) {
            const float s  = ps[0][col]  + ps[1][col]  + ps[2][col]  + ps[3][col];
            const float s2 = ps2[0][col] + ps2[1][col] + ps2[2][col] + ps2[3][col];
            atomicAdd(&stats[c0 + col], s);
            atomicAdd(&stats[CC + c0 + col], s2);
        }
    }
}

// ---------------------------------------------------------------------------
// Kernel 7: BatchNorm (batch stats) + residual add.
// ---------------------------------------------------------------------------
__global__ __launch_bounds__(256) void bn_residual_kernel(
    const float* __restrict__ Wy, const float* __restrict__ stats,
    const float* __restrict__ gamma, const float* __restrict__ beta,
    const float* __restrict__ x, float* __restrict__ out)
{
    const size_t i = ((size_t)blockIdx.x * 256 + threadIdx.x) * 4;
    const int c = (int)((i >> 12) & (CC - 1));      // N = 4096 = 2^12

    const float invn = 1.0f / (float)(BB * NN);
    const float mu = stats[c] * invn;
    const float var = stats[CC + c] * invn - mu * mu;
    const float scale = rsqrtf(var + BN_EPS) * gamma[c];
    const float shift = beta[c] - mu * scale;

    const float4 wy = *(const float4*)(Wy + i);
    const float4 xx = *(const float4*)(x + i);
    float4 o;
    o.x = wy.x * scale + shift + xx.x;
    o.y = wy.y * scale + shift + xx.y;
    o.z = wy.z * scale + shift + xx.z;
    o.w = wy.w * scale + shift + xx.w;
    *(float4*)(out + i) = o;
}

// ---------------------------------------------------------------------------
extern "C" void kernel_launch(void* const* d_in, const int* in_sizes, int n_in,
                              void* d_out, int out_size, void* d_ws, size_t ws_size,
                              hipStream_t stream) {
    (void)in_sizes; (void)n_in; (void)out_size; (void)ws_size;

    const float* x          = (const float*)d_in[0];
    // d_in[1] = mask (unused by the reference)
    const float* mask_prev  = (const float*)d_in[2];
    const float* nl_prev    = (const float*)d_in[3];
    const float* g_w        = (const float*)d_in[4];
    const float* g_b        = (const float*)d_in[5];
    const float* theta_w    = (const float*)d_in[6];
    const float* theta_b    = (const float*)d_in[7];
    const float* phi_w      = (const float*)d_in[8];
    const float* phi_b      = (const float*)d_in[9];
    const float* W_w        = (const float*)d_in[10];
    const float* W_b        = (const float*)d_in[11];
    const float* bn_gamma   = (const float*)d_in[12];
    const float* bn_beta    = (const float*)d_in[13];
    const float* lamb       = (const float*)d_in[14];
    const float* lamb_sm    = (const float*)d_in[15];
    float* out = (float*)d_out;

    // Workspace layout (byte offsets); fh aliases dead f, Wy aliases dead fblend.
    char* ws = (char*)d_ws;
    _Float16* theta  = (_Float16*)(ws);                        //  8 MB (B,N,Ci)
    _Float16* gbuf   = (_Float16*)(ws + 8388608);              //  2 MB (B,Ci,M)
    _Float16* phibuf = (_Float16*)(ws + 10485760);             //  2 MB (B,M,Ci)
    float*    fbuf   = (float*)   (ws + 12582912);             // 134 MB (B,N,M)
    float*    fblend = (float*)   (ws + 146800640);            // 134 MB (B,N,M)
    _Float16* fh     = (_Float16*)fbuf;                        // alias (67 MB)
    float*    Wy     = fblend;                                 // alias (33.5 MB)
    _Float16* ybuf   = (_Float16*)(ws + 281018368);            //  8 MB (B,N,Ci)
    float*    stats  = (float*)   (ws + 289406976);            //  2 KB

    const dim3 blk(256);

    proj_kernel<<<dim3(NN / 128, CI / 64, BB * 3), blk, 0, stream>>>(
        x, g_w, g_b, theta_w, theta_b, phi_w, phi_b, theta, gbuf, phibuf);

    f_gemm_kernel<<<dim3(MM / 128, NN / 128, BB), blk, 0, stream>>>(
        theta, phibuf, fbuf);

    smooth_kernel<<<dim3((BB * NN * MM) / 256), blk, 0, stream>>>(
        fbuf, lamb_sm, fblend);

    softmax_kernel<<<dim3(BB * NN), blk, 0, stream>>>(
        fblend, nl_prev, mask_prev, lamb, fh);

    y_gemm_kernel<<<dim3(CI / 64, NN / 128, BB), blk, 0, stream>>>(
        fh, gbuf, ybuf);

    hipMemsetAsync(stats, 0, 2 * CC * sizeof(float), stream);

    wy_gemm_kernel<<<dim3(CC / 64, NN / 128, BB), blk, 0, stream>>>(
        ybuf, W_w, W_b, Wy, stats);

    bn_residual_kernel<<<dim3((BB * CC * NN) / (256 * 4)), blk, 0, stream>>>(
        Wy, stats, bn_gamma, bn_beta, x, out);
}
